// DINOLoss_21363167330883
// MI455X (gfx1250) — matrix operand order repaired and checked by reading, not compiled, not run on hardware
//
#include <hip/hip_runtime.h>
#include <hip/hip_bf16.h>

typedef __attribute__((ext_vector_type(16))) __bf16 v16bf;
typedef __attribute__((ext_vector_type(8)))  __bf16 v8bf;
typedef __attribute__((ext_vector_type(8)))  float  v8f;

union BFrag { v16bf v; struct { v8bf lo, hi; } p; };

#define D_DIM 256

// ---------------------------------------------------------------- fp32 -> bf16
__global__ void k_convert(const float* __restrict__ src, __bf16* __restrict__ dst, long n) {
  long i = (long)blockIdx.x * blockDim.x + threadIdx.x;
  long stride = (long)gridDim.x * blockDim.x;
  for (; i < n; i += stride) dst[i] = (__bf16)src[i];
}

// -------------------------------------------------- per-class centroid + norm
__global__ void k_centroid(const float* __restrict__ queue, const int* __restrict__ label,
                           const int* __restrict__ numClassesPtr,
                           float* __restrict__ cent, float* __restrict__ counts, int K) {
  int c = blockIdx.x;
  int C = *numClassesPtr;
  if (c >= C) return;
  int d = threadIdx.x;                      // d in [0,256)
  float acc = 0.f; int cnt = 0;
  for (int k = 0; k < K; ++k) {
    if (label[k] == c) { acc += queue[(long)k * D_DIM + d]; cnt++; }
  }
  float ctr = acc / (float)cnt;             // segment mean
  __shared__ float red[256];
  red[d] = ctr * ctr;
  __syncthreads();
  for (int s = 128; s > 0; s >>= 1) { if (d < s) red[d] += red[d + s]; __syncthreads(); }
  float nrm = fmaxf(sqrtf(red[0]), 1e-12f);
  cent[(long)c * D_DIM + d] = ctr / nrm;
  if (d == 0) counts[c] = (float)cnt;
}

// ------------------------------------------------------- pseudo-label argmax
__global__ void k_pseudo(const float* __restrict__ batch, const float* __restrict__ cent,
                         const int* __restrict__ numClassesPtr, int* __restrict__ pl) {
  int b = blockIdx.x;
  int C = *numClassesPtr;
  __shared__ float row[256];
  __shared__ float sv[256];
  __shared__ int   si[256];
  int t = threadIdx.x;
  row[t] = batch[(long)b * D_DIM + t];
  __syncthreads();
  float best = -1e30f;
  if (t < C) {
    const float* cp = cent + (long)t * D_DIM;
    float s = 0.f;
    for (int i = 0; i < D_DIM; ++i) s = fmaf(row[i], cp[i], s);
    best = s;
  }
  sv[t] = best; si[t] = t;
  __syncthreads();
  for (int s = 128; s > 0; s >>= 1) {
    if (t < s) {
      float v2 = sv[t + s]; int i2 = si[t + s];
      if (v2 > sv[t] || (v2 == sv[t] && i2 < si[t])) { sv[t] = v2; si[t] = i2; }
    }
    __syncthreads();
  }
  if (t == 0) pl[b] = si[0];
}

// ------------------------------------------------------------------- zeroing
__global__ void k_zero(float* __restrict__ p, int n) {
  int i = blockIdx.x * 256 + threadIdx.x;
  if (i < n) p[i] = 0.f;
}

// ----------------- fused WMMA GEMM (batch @ queue^T) + sqrt + masked row sums
// block = 256 threads = 8 waves. Each wave owns a distinct 16-row A tile
// (block covers 128 rows); ALL waves walk the SAME 16-col queue-tile sequence,
// so the 8 waves hit identical B addresses and the WGP$ serves 7 of 8 —
// memory-side B traffic drops 8x vs per-wave column striding.
// Software-pipelined B (+label) register double buffer; two WMMA chains.
__launch_bounds__(256)
__global__ void k_mae(const __bf16* __restrict__ batchBF, const __bf16* __restrict__ queueBF,
                      const int* __restrict__ label, const int* __restrict__ pl,
                      float* __restrict__ S_tot, float* __restrict__ S_msk,
                      int B, int K) {
  const int lane = threadIdx.x & 31;
  const int wave = threadIdx.x >> 5;
  const int half = lane >> 4;        // 0: lanes 0-15, 1: lanes 16-31
  const int ln   = lane & 15;
  const int m0   = (blockIdx.x * 8 + wave) * 16;   // this wave's 16 A rows
  if (m0 >= B) return;                              // no barriers: safe to exit

  const int kPer = K / gridDim.y;    // multiple of 16 (host guarantees)
  const int kBeg = blockIdx.y * kPer;
  const int kEnd = kBeg + kPer;

  // A fragments for the full D=256 (16x32 bf16 A layout, ISA 7.12.2):
  // lane l (half h) holds row m0+ln, K = ks*32 + h*8 .. +8 and +16.
  BFrag a[8];
  {
    const __bf16* arow = batchBF + (long)(m0 + ln) * D_DIM;
    #pragma unroll
    for (int ks = 0; ks < 8; ++ks) {
      int klo = ks * 32 + half * 8;
      a[ks].p.lo = *(const v8bf*)(arow + klo);
      a[ks].p.hi = *(const v8bf*)(arow + klo + 16);
    }
  }

  // C/D layout: lane holds column n0+ln, rows m0 + half*8 + v (v=0..7).
  int plv[8];
  #pragma unroll
  for (int v = 0; v < 8; ++v) plv[v] = pl[m0 + half * 8 + v];

  float accT[8], accM[8];
  #pragma unroll
  for (int v = 0; v < 8; ++v) { accT[v] = 0.f; accM[v] = 0.f; }

  // per-lane B base: queue row ln of the tile, K = ks*32 + h*8 .. +8 and +16 per frag
  const __bf16* qbase = queueBF + (long)ln * D_DIM + half * 8;

  // load B frags AND this tile's column label (pipelined together)
  auto loadB = [&](BFrag* b, int& lab, int n0) {
    lab = label[n0 + ln];
    const __bf16* brow = qbase + (long)n0 * D_DIM;
    __builtin_prefetch(brow + 4 * 16 * D_DIM, 0, 1);   // 4 tiles ahead -> L2
    #pragma unroll
    for (int ks = 0; ks < 8; ++ks) {
      b[ks].p.lo = *(const v8bf*)(brow + ks * 32);
      b[ks].p.hi = *(const v8bf*)(brow + ks * 32 + 16);
    }
  };

  auto process = [&](const BFrag* b, int lab) {
    v8f c0 = {}, c1 = {};
    #pragma unroll
    for (int ks = 0; ks < 8; ks += 2) {
      c0 = __builtin_amdgcn_wmma_f32_16x16x32_bf16(false, a[ks].v,     false, b[ks].v,
                                                   (short)0, c0, false, false);
      c1 = __builtin_amdgcn_wmma_f32_16x16x32_bf16(false, a[ks + 1].v, false, b[ks + 1].v,
                                                   (short)0, c1, false, false);
    }
    #pragma unroll
    for (int v = 0; v < 8; ++v) {
      float s = c0[v] + c1[v];
      // arg >= 1e-6: never denormal -> raw v_sqrt_f32, no range fixup needed
      float mae = __builtin_amdgcn_sqrtf(fmaf(-2.f, s, 2.0f + 1e-6f));
      accT[v] += mae;
      if (plv[v] == lab) accM[v] += mae;
    }
  };

  // all waves iterate the same column tiles: stride 16
  int n = kBeg;
  {
    BFrag b0[8], b1[8];
    int lab0, lab1;
    loadB(b0, lab0, n);
    while (true) {
      int nn = n + 16;
      if (nn < kEnd) loadB(b1, lab1, nn);  // overlap with b0's WMMAs
      process(b0, lab0);
      if (nn >= kEnd) break;
      n = nn;
      nn = n + 16;
      if (nn < kEnd) loadB(b0, lab0, nn);  // overlap with b1's WMMAs
      process(b1, lab1);
      if (nn >= kEnd) break;
      n = nn;
    }
  }

  // reduce over the 16 columns held by each 16-lane group, then atomics per row
  #pragma unroll
  for (int v = 0; v < 8; ++v) {
    float t = accT[v], m = accM[v];
    #pragma unroll
    for (int off = 1; off < 16; off <<= 1) {
      t += __shfl_xor(t, off, 32);
      m += __shfl_xor(m, off, 32);
    }
    if (ln == 0) {
      atomicAdd(&S_tot[m0 + half * 8 + v], t);
      atomicAdd(&S_msk[m0 + half * 8 + v], m);
    }
  }
}

// -------------------------------------------------------------- final scalar
__global__ void k_final(const float* __restrict__ S_tot, const float* __restrict__ S_msk,
                        const int* __restrict__ pl, const float* __restrict__ counts,
                        float* __restrict__ out, int B, int K) {
  __shared__ float r1[256], r2[256];
  int t = threadIdx.x;
  float s1 = 0.f, s2 = 0.f;
  for (int b = t; b < B; b += 256) {
    float cnt = counts[pl[b]];
    float msk = S_msk[b], tot = S_tot[b];
    s1 += msk / (cnt + 1e-6f);
    s2 += (tot - msk) / ((float)K - cnt + 1e-6f);
  }
  r1[t] = s1; r2[t] = s2;
  __syncthreads();
  for (int s = 128; s > 0; s >>= 1) {
    if (t < s) { r1[t] += r1[t + s]; r2[t] += r2[t + s]; }
    __syncthreads();
  }
  if (t == 0) out[0] = r1[0] / (float)B + 2.0f - r2[0] / (float)B;
}

extern "C" void kernel_launch(void* const* d_in, const int* in_sizes, int n_in,
                              void* d_out, int out_size, void* d_ws, size_t ws_size,
                              hipStream_t stream) {
  (void)n_in; (void)out_size; (void)ws_size;
  const float* batch = (const float*)d_in[0];   // [B, 256] f32, L2-normalized
  const float* queue = (const float*)d_in[1];   // [K, 256] f32, L2-normalized
  const int*   label = (const int*)d_in[2];     // [K] int32
  const int*   numC  = (const int*)d_in[3];     // device scalar

  const int B = in_sizes[0] / D_DIM;
  const int K = in_sizes[2];

  // workspace layout (256B aligned regions)
  char* ws = (char*)d_ws;
  size_t o = 0;
  auto rnd = [](size_t x) { return (x + 255) & ~(size_t)255; };
  __bf16* batchBF = (__bf16*)(ws + o); o += rnd((size_t)B * D_DIM * 2);
  __bf16* queueBF = (__bf16*)(ws + o); o += rnd((size_t)K * D_DIM * 2);
  float*  cent    = (float*)(ws + o);  o += rnd((size_t)256 * D_DIM * 4);
  float*  counts  = (float*)(ws + o);  o += 1024;
  int*    pl      = (int*)(ws + o);    o += rnd((size_t)B * 4);
  float*  S_tot   = (float*)(ws + o);  o += rnd((size_t)B * 4);
  float*  S_msk   = (float*)(ws + o);

  // phase 0: zero row accumulators
  k_zero<<<dim3((B + 255) / 256), 256, 0, stream>>>(S_tot, B);
  k_zero<<<dim3((B + 255) / 256), 256, 0, stream>>>(S_msk, B);

  // phase 1: bf16 conversions (independent of phases 2-3)
  k_convert<<<dim3(1024), 256, 0, stream>>>(batch, batchBF, (long)B * D_DIM);
  k_convert<<<dim3(4096), 256, 0, stream>>>(queue, queueBF, (long)K * D_DIM);

  // phase 2: per-class centroids (normalized) + class histogram
  k_centroid<<<dim3(256), 256, 0, stream>>>(queue, label, numC, cent, counts, K);

  // phase 3: pseudo-labels via argmax over centroid similarities
  k_pseudo<<<dim3(B), 256, 0, stream>>>(batch, cent, numC, pl);

  // phase 4: fused WMMA GEMM + MAE + masked row reductions
  // block covers 128 rows (8 waves x 16); K split 16 ways for occupancy
  int rowBlocks = (B + 127) / 128;
  int nsplits = 16;
  if (K % (16 * nsplits) != 0) nsplits = 1;
  k_mae<<<dim3(rowBlocks, nsplits), 256, 0, stream>>>(batchBF, queueBF, label, pl,
                                                      S_tot, S_msk, B, K);

  // phase 5: final scalar loss
  k_final<<<dim3(1), 256, 0, stream>>>(S_tot, S_msk, pl, counts, (float*)d_out, B, K);
}
